// Attention_67233418051996
// MI455X (gfx1250) — matrix operand order repaired and checked by reading, not compiled, not run on hardware
//
#include <hip/hip_runtime.h>
#include <hip/hip_bf16.h>
#include <math.h>

typedef __bf16 bf16t;
typedef __attribute__((ext_vector_type(16))) __bf16 v16bf;
typedef __attribute__((ext_vector_type(8)))  __bf16 v8bf;
typedef __attribute__((ext_vector_type(4)))  __bf16 v4bf;
typedef __attribute__((ext_vector_type(8)))  float  v8f;

#define DEV __device__ __forceinline__

// ---------------- WMMA helpers (CDNA5 wave32, 16x16x32 bf16 -> f32) ----------

DEV v8f wmma_bf16(v16bf a, v16bf b, v8f c) {
  return __builtin_amdgcn_wmma_f32_16x16x32_bf16(false, a, false, b, (short)0, c,
                                                 false, false);
}

// A-matrix 16x32 bf16 fragment (works on global or LDS pointers).
DEV v16bf load_a_frag(const bf16t* row, int kbase, int lane) {
  int off = (lane >> 4) << 3;                      // 0 or 8
  v8bf c1 = *(const v8bf*)(row + kbase + off);
  v8bf c2 = *(const v8bf*)(row + kbase + off + 16);
  return __builtin_shufflevector(c1, c2, 0,1,2,3,4,5,6,7,8,9,10,11,12,13,14,15);
}

// B-matrix 32x16 bf16 fragment, B[k][n] = W[n][k], W row-major.
DEV v16bf load_b_frag(const bf16t* wrow, int kbase, int lane) {
  int off = (lane >> 4) << 3;                      // 0 or 8
  v8bf c1 = *(const v8bf*)(wrow + kbase + off);
  v8bf c2 = *(const v8bf*)(wrow + kbase + off + 16);
  return __builtin_shufflevector(c1, c2, 0,1,2,3,4,5,6,7,8,9,10,11,12,13,14,15);
}

DEV void wmma8(v8f* acc, v16bf a0, v16bf a1,
               v16bf b0, v16bf b1, v16bf b2, v16bf b3) {
  acc[0] = wmma_bf16(a0, b0, acc[0]);
  acc[1] = wmma_bf16(a0, b1, acc[1]);
  acc[2] = wmma_bf16(a0, b2, acc[2]);
  acc[3] = wmma_bf16(a0, b3, acc[3]);
  acc[4] = wmma_bf16(a1, b0, acc[4]);
  acc[5] = wmma_bf16(a1, b1, acc[5]);
  acc[6] = wmma_bf16(a1, b2, acc[6]);
  acc[7] = wmma_bf16(a1, b3, acc[7]);
}

// ---------------- CDNA5 async global->LDS copy (ASYNCcnt tracked) -----------

// 16 bytes/lane; LDS byte address = low 32 bits of the flat shared pointer.
DEV void async_a_chunk(const bf16t* gsrc, bf16t* ldst) {
  uint32_t laddr = (uint32_t)(uintptr_t)ldst;
  asm volatile("global_load_async_to_lds_b128 %0, %1, off"
               :: "v"(laddr), "v"(gsrc) : "memory");
}
DEV void wait_async_barrier() {
  asm volatile("s_wait_asynccnt 0x0" ::: "memory");
  __syncthreads();
}

// LDS-staged GEMM core: block = 8 waves sharing a 32-row A tile, each wave owns
// a 64-col B slice. Stage = 32 rows x 128 K (8 KB), double buffered.
// Stager(kbase, lbuf) issues this wave's 2 async b128 copies for one stage.
template<int KDIM, typename Stager>
DEV void gemm_lds_loop(Stager stage, const bf16t* w0, const bf16t* w1,
                       const bf16t* w2, const bf16t* w3,
                       bf16t* lds0, int lane, int ln, v8f* acc) {
  constexpr int S = KDIM / 128;
  stage(0, lds0);
  v16bf b0 = load_b_frag(w0, 0, lane), b1 = load_b_frag(w1, 0, lane);
  v16bf b2 = load_b_frag(w2, 0, lane), b3 = load_b_frag(w3, 0, lane);
  for (int s = 0; s < S; ++s) {
    wait_async_barrier();                          // stage s resident in LDS
    bf16t* cur = lds0 + (s & 1) * (32 * 128);
    if (s + 1 < S) stage((s + 1) * 128, lds0 + ((s + 1) & 1) * (32 * 128));
    #pragma unroll
    for (int ks = 0; ks < 4; ++ks) {
      int kt = s * 128 + ks * 32;
      int ktn = (kt + 32 < KDIM) ? kt + 32 : 0;    // wrap: last reload unused
      v16bf a0 = load_a_frag(cur + ln * 128, ks * 32, lane);
      v16bf a1 = load_a_frag(cur + (16 + ln) * 128, ks * 32, lane);
      v16bf nb0 = load_b_frag(w0, ktn, lane), nb1 = load_b_frag(w1, ktn, lane);
      v16bf nb2 = load_b_frag(w2, ktn, lane), nb3 = load_b_frag(w3, ktn, lane);
      wmma8(acc, a0, a1, b0, b1, b2, b3);
      b0 = nb0; b1 = nb1; b2 = nb2; b3 = nb3;
    }
  }
}

DEV float half_max16(float x) {
  #pragma unroll
  for (int m = 1; m < 16; m <<= 1) x = fmaxf(x, __shfl_xor(x, m, 32));
  return x;
}
DEV float half_sum16(float x) {
  #pragma unroll
  for (int m = 1; m < 16; m <<= 1) x += __shfl_xor(x, m, 32);
  return x;
}

// ---------------- conversion kernels ----------------

__global__ void cvt_bf16x4_kernel(const float4* __restrict__ in,
                                  v4bf* __restrict__ out, int n4) {
  int i = blockIdx.x * blockDim.x + threadIdx.x;
  if (i < n4) {
    float4 f = in[i];
    v4bf o;
    o[0] = (bf16t)f.x; o[1] = (bf16t)f.y; o[2] = (bf16t)f.z; o[3] = (bf16t)f.w;
    out[i] = o;
  }
}

// conv_w (O=512, I=512, 2, 2) -> Wc[o][p*512 + i] with p = kh*2+kw
__global__ void rearrange_convw_kernel(const float* __restrict__ cw,
                                       bf16t* __restrict__ out) {
  int i = blockIdx.x * blockDim.x + threadIdx.x;   // 512*2048 threads
  int o = i >> 11, kk = i & 2047;
  int p = kk >> 9, c = kk & 511;
  out[i] = (bf16t)cw[(o * 512 + c) * 4 + p];
}

// ---------------- GEMM: Q = x @ Wq^T, scatter to qh[b][h][n][d] -------------

__global__ __launch_bounds__(256) void gemm_q_kernel(
    const bf16t* __restrict__ xh, const bf16t* __restrict__ Wq,
    bf16t* __restrict__ qh) {
  __shared__ __attribute__((aligned(16))) bf16t Atile[2 * 32 * 128];
  int lane = threadIdx.x & 31, ln = lane & 15, half = lane >> 4;
  int wid = threadIdx.x >> 5;
  int m0 = blockIdx.x << 5;                        // block shares 32 A rows
  int ntile = wid, n0 = ntile << 6;                // wave owns 64 cols
  auto stager = [=](int kbase, bf16t* lbuf) {
    #pragma unroll
    for (int i2 = 0; i2 < 2; ++i2) {
      int chunk = ((wid * 2 + i2) << 5) + lane;    // 512 chunks of 8 elems
      int row = chunk >> 4, kc = (chunk & 15) << 3;
      async_a_chunk(xh + (size_t)(m0 + row) * 512 + kbase + kc,
                    lbuf + row * 128 + kc);
    }
  };
  v8f acc[8] = {};
  gemm_lds_loop<512>(stager,
                     Wq + (size_t)(n0 + ln) * 512,
                     Wq + (size_t)(n0 + 16 + ln) * 512,
                     Wq + (size_t)(n0 + 32 + ln) * 512,
                     Wq + (size_t)(n0 + 48 + ln) * 512, Atile, lane, ln, acc);
  int h = ntile;                                   // 64-wide tile == one head
  #pragma unroll
  for (int mi = 0; mi < 2; ++mi)
    #pragma unroll
    for (int t = 0; t < 4; ++t)
      #pragma unroll
      for (int v = 0; v < 8; ++v) {
        int m = m0 + (mi << 4) + v + (half << 3);
        int b = m >> 12, n = m & 4095;
        int d = (t << 4) + ln;
        qh[(((size_t)(b * 8 + h) * 4096 + n) << 6) + d] = (bf16t)acc[(mi << 2) + t][v];
      }
}

// ---------------- GEMM: conv-as-GEMM, fp32 out + bias -----------------------

DEV const bf16t* conv_arow(const bf16t* xh, int m, int kt) {
  int b = m >> 10, mloc = m & 1023;
  int oy = mloc >> 5, ox = mloc & 31;
  int p = kt >> 9;                                 // patch position 0..3
  int iy = (oy << 1) + (p >> 1), ix = (ox << 1) + (p & 1);
  return xh + ((size_t)b * 4096 + iy * 64 + ix) * 512;
}

__global__ __launch_bounds__(256) void gemm_conv_kernel(
    const bf16t* __restrict__ xh, const bf16t* __restrict__ Wc,
    const float* __restrict__ conv_b, float* __restrict__ ctxf) {
  __shared__ __attribute__((aligned(16))) bf16t Atile[2 * 32 * 128];
  int lane = threadIdx.x & 31, ln = lane & 15, half = lane >> 4;
  int wid = threadIdx.x >> 5;
  int m0 = blockIdx.x << 5;
  int n0 = wid << 6;
  auto stager = [=](int kbase, bf16t* lbuf) {
    #pragma unroll
    for (int i2 = 0; i2 < 2; ++i2) {
      int chunk = ((wid * 2 + i2) << 5) + lane;
      int row = chunk >> 4, kc = (chunk & 15) << 3;
      const bf16t* ar = conv_arow(xh, m0 + row, kbase);
      async_a_chunk(ar + (kbase & 511) + kc, lbuf + row * 128 + kc);
    }
  };
  v8f acc[8] = {};
  gemm_lds_loop<2048>(stager,
                      Wc + (size_t)(n0 + ln) * 2048,
                      Wc + (size_t)(n0 + 16 + ln) * 2048,
                      Wc + (size_t)(n0 + 32 + ln) * 2048,
                      Wc + (size_t)(n0 + 48 + ln) * 2048, Atile, lane, ln, acc);
  #pragma unroll
  for (int mi = 0; mi < 2; ++mi)
    #pragma unroll
    for (int t = 0; t < 4; ++t)
      #pragma unroll
      for (int v = 0; v < 8; ++v) {
        int mg = m0 + (mi << 4) + v + (half << 3);
        int col = n0 + (t << 4) + ln;
        ctxf[(size_t)mg * 512 + col] = acc[(mi << 2) + t][v] + conv_b[col];
      }
}

// ---------------- LayerNorm (512 wide) -> bf16 ------------------------------

__global__ __launch_bounds__(128) void ln_kernel(
    const float* __restrict__ ctxf, const float* __restrict__ lnw,
    const float* __restrict__ lnb, bf16t* __restrict__ ctxh) {
  int r = blockIdx.x, tid = threadIdx.x;
  float4 v4 = ((const float4*)(ctxf + (size_t)r * 512))[tid];
  float s  = v4.x + v4.y + v4.z + v4.w;
  float s2 = v4.x * v4.x + v4.y * v4.y + v4.z * v4.z + v4.w * v4.w;
  #pragma unroll
  for (int m = 1; m < 32; m <<= 1) { s += __shfl_xor(s, m, 32); s2 += __shfl_xor(s2, m, 32); }
  __shared__ float sa[4], sb[4];
  int w = tid >> 5;
  if ((tid & 31) == 0) { sa[w] = s; sb[w] = s2; }
  __syncthreads();
  float S  = sa[0] + sa[1] + sa[2] + sa[3];
  float S2 = sb[0] + sb[1] + sb[2] + sb[3];
  float mu  = S * (1.f / 512.f);
  float var = S2 * (1.f / 512.f) - mu * mu;
  float inv = rsqrtf(var + 1e-5f);
  int c = tid << 2;
  bf16t* orow = ctxh + (size_t)r * 512 + c;
  orow[0] = (bf16t)((v4.x - mu) * inv * lnw[c + 0] + lnb[c + 0]);
  orow[1] = (bf16t)((v4.y - mu) * inv * lnw[c + 1] + lnb[c + 1]);
  orow[2] = (bf16t)((v4.z - mu) * inv * lnw[c + 2] + lnb[c + 2]);
  orow[3] = (bf16t)((v4.w - mu) * inv * lnw[c + 3] + lnb[c + 3]);
}

// ---------------- GEMM: KV = ctx @ Wkv^T; K->[b][h][m][d], V->[b][h][d][m] --

__global__ __launch_bounds__(256) void gemm_kv_kernel(
    const bf16t* __restrict__ ctxh, const bf16t* __restrict__ Wkv,
    bf16t* __restrict__ khp, bf16t* __restrict__ vTp) {
  __shared__ __attribute__((aligned(16))) bf16t Atile[2 * 32 * 128];
  int lane = threadIdx.x & 31, ln = lane & 15, half = lane >> 4;
  int wid = threadIdx.x >> 5;
  int m0 = (blockIdx.x >> 1) << 5;                 // 2 blocks per A tile
  int ntile = ((blockIdx.x & 1) << 3) + wid;       // 16 ntiles
  int n0 = ntile << 6;
  auto stager = [=](int kbase, bf16t* lbuf) {
    #pragma unroll
    for (int i2 = 0; i2 < 2; ++i2) {
      int chunk = ((wid * 2 + i2) << 5) + lane;
      int row = chunk >> 4, kc = (chunk & 15) << 3;
      async_a_chunk(ctxh + (size_t)(m0 + row) * 512 + kbase + kc,
                    lbuf + row * 128 + kc);
    }
  };
  v8f acc[8] = {};
  gemm_lds_loop<512>(stager,
                     Wkv + (size_t)(n0 + ln) * 512,
                     Wkv + (size_t)(n0 + 16 + ln) * 512,
                     Wkv + (size_t)(n0 + 32 + ln) * 512,
                     Wkv + (size_t)(n0 + 48 + ln) * 512, Atile, lane, ln, acc);
  bool isK = (n0 < 512);                           // uniform per wave
  #pragma unroll
  for (int mi = 0; mi < 2; ++mi)
    #pragma unroll
    for (int t = 0; t < 4; ++t)
      #pragma unroll
      for (int v = 0; v < 8; ++v) {
        int m = m0 + (mi << 4) + v + (half << 3);
        int b = m >> 10, ml = m & 1023;
        int col = n0 + (t << 4) + ln;
        float val = acc[(mi << 2) + t][v];
        if (isK) {
          int h = col >> 6, d = col & 63;
          khp[(((size_t)(b * 8 + h) * 1024 + ml) << 6) + d] = (bf16t)val;
        } else {
          int c2 = col - 512, h = c2 >> 6, d = c2 & 63;
          vTp[(((size_t)(b * 8 + h) * 64 + d) << 10) + ml] = (bf16t)val;
        }
      }
}

// ---------------- Flash attention: 1 wave = 16 queries, full DH=64 ----------
// Block = one (b,h) with 8 consecutive query tiles: all waves stream the same
// K/V tiles (L0/L2 reuse).

__global__ __launch_bounds__(256) void attn_kernel(
    const bf16t* __restrict__ qh, const bf16t* __restrict__ khp,
    const bf16t* __restrict__ vTp, const float* __restrict__ mask,
    bf16t* __restrict__ o2) {
  __shared__ __attribute__((aligned(16))) bf16t Pst[8][16 * 32];
  const float SCALE = 0.125f;                      // 64^-0.5
  const float NEG_INF = -__builtin_inff();
  int lane = threadIdx.x & 31, ln = lane & 15, half = lane >> 4;
  int wid = threadIdx.x >> 5;
  int bh = blockIdx.x >> 5;                        // 64 (b,h) pairs
  int qt = ((blockIdx.x & 31) << 3) + wid;         // 256 query tiles
  int b = bh >> 3, h = bh & 7;

  const bf16t* qrow = qh + (((size_t)bh * 4096) + (qt << 4) + ln) * 64;
  v16bf aq0 = load_a_frag(qrow, 0, lane);
  v16bf aq1 = load_a_frag(qrow, 32, lane);

  v8f acc0 = {}, acc1 = {}, acc2 = {}, acc3 = {};
  float mrow[8], lrow[8];
  #pragma unroll
  for (int v = 0; v < 8; ++v) { mrow[v] = NEG_INF; lrow[v] = 0.f; }

  const bf16t* kb = khp + ((size_t)bh << 10) * 64; // [1024][64]
  const bf16t* vb = vTp + ((size_t)bh << 6) * 1024;// [64][1024]
  const float* mk = mask + b * 1024;
  bf16t* myP = &Pst[wid][0];

  v16bf bk0 = load_b_frag(kb + (size_t)ln * 64, 0, lane);
  v16bf bk1 = load_b_frag(kb + (size_t)ln * 64, 32, lane);
  v16bf bk2 = load_b_frag(kb + (size_t)(16 + ln) * 64, 0, lane);
  v16bf bk3 = load_b_frag(kb + (size_t)(16 + ln) * 64, 32, lane);

  for (int j0 = 0; j0 < 1024; j0 += 32) {
    v8f z = {};
    v8f s0 = wmma_bf16(aq0, bk0, z);
    s0     = wmma_bf16(aq1, bk1, s0);
    v8f s1 = wmma_bf16(aq0, bk2, z);
    s1     = wmma_bf16(aq1, bk3, s1);

    // V loads for this tile + K loads for next; softmax VALU hides them
    v16bf bv0 = load_b_frag(vb + (size_t)(ln)      * 1024 + j0, 0, lane);
    v16bf bv1 = load_b_frag(vb + (size_t)(16 + ln) * 1024 + j0, 0, lane);
    v16bf bv2 = load_b_frag(vb + (size_t)(32 + ln) * 1024 + j0, 0, lane);
    v16bf bv3 = load_b_frag(vb + (size_t)(48 + ln) * 1024 + j0, 0, lane);
    int jn = (j0 + 32) & 1023;
    v16bf nk0 = load_b_frag(kb + (size_t)(jn + ln) * 64, 0, lane);
    v16bf nk1 = load_b_frag(kb + (size_t)(jn + ln) * 64, 32, lane);
    v16bf nk2 = load_b_frag(kb + (size_t)(jn + 16 + ln) * 64, 0, lane);
    v16bf nk3 = load_b_frag(kb + (size_t)(jn + 16 + ln) * 64, 32, lane);

    float mv0 = mk[j0 + ln];
    float mv1 = mk[j0 + 16 + ln];
    #pragma unroll
    for (int v = 0; v < 8; ++v) {
      float a = (mv0 != 0.f) ? s0[v] * SCALE : NEG_INF;
      float c = (mv1 != 0.f) ? s1[v] * SCALE : NEG_INF;
      float rm = half_max16(fmaxf(a, c));
      float mn = fmaxf(mrow[v], rm);
      float mne = (mn == NEG_INF) ? 0.f : mn;      // avoid inf-inf NaN
      float scale = __expf(mrow[v] - mne);
      float p0 = __expf(a - mne);
      float p1 = __expf(c - mne);
      lrow[v] = lrow[v] * scale + half_sum16(p0 + p1);
      mrow[v] = mn;
      acc0[v] *= scale; acc1[v] *= scale; acc2[v] *= scale; acc3[v] *= scale;
      int row = v + (half << 3);
      myP[row * 32 + ln]      = (bf16t)p0;         // stage P (C-layout -> LDS)
      myP[row * 32 + 16 + ln] = (bf16t)p1;
    }
    v16bf ap = load_a_frag(myP + ln * 32, 0, lane);
    acc0 = wmma_bf16(ap, bv0, acc0);
    acc1 = wmma_bf16(ap, bv1, acc1);
    acc2 = wmma_bf16(ap, bv2, acc2);
    acc3 = wmma_bf16(ap, bv3, acc3);
    bk0 = nk0; bk1 = nk1; bk2 = nk2; bk3 = nk3;
  }

  #pragma unroll
  for (int v = 0; v < 8; ++v) {
    float inv = 1.f / lrow[v];
    int row = v + (half << 3);
    bf16t* orow = o2 + ((size_t)b * 4096 + (qt << 4) + row) * 512 + h * 64;
    orow[ln]      = (bf16t)(acc0[v] * inv);
    orow[16 + ln] = (bf16t)(acc1[v] * inv);
    orow[32 + ln] = (bf16t)(acc2[v] * inv);
    orow[48 + ln] = (bf16t)(acc3[v] * inv);
  }
}

// ---------------- GEMM: out = o2 @ Wp^T + bp (fp32 output) ------------------

__global__ __launch_bounds__(256) void gemm_out_kernel(
    const bf16t* __restrict__ o2, const bf16t* __restrict__ Wp,
    const float* __restrict__ bp, float* __restrict__ out) {
  __shared__ __attribute__((aligned(16))) bf16t Atile[2 * 32 * 128];
  int lane = threadIdx.x & 31, ln = lane & 15, half = lane >> 4;
  int wid = threadIdx.x >> 5;
  int m0 = blockIdx.x << 5;
  int n0 = wid << 6;
  auto stager = [=](int kbase, bf16t* lbuf) {
    #pragma unroll
    for (int i2 = 0; i2 < 2; ++i2) {
      int chunk = ((wid * 2 + i2) << 5) + lane;
      int row = chunk >> 4, kc = (chunk & 15) << 3;
      async_a_chunk(o2 + (size_t)(m0 + row) * 512 + kbase + kc,
                    lbuf + row * 128 + kc);
    }
  };
  v8f acc[8] = {};
  gemm_lds_loop<512>(stager,
                     Wp + (size_t)(n0 + ln) * 512,
                     Wp + (size_t)(n0 + 16 + ln) * 512,
                     Wp + (size_t)(n0 + 32 + ln) * 512,
                     Wp + (size_t)(n0 + 48 + ln) * 512, Atile, lane, ln, acc);
  #pragma unroll
  for (int mi = 0; mi < 2; ++mi)
    #pragma unroll
    for (int t = 0; t < 4; ++t)
      #pragma unroll
      for (int v = 0; v < 8; ++v) {
        int m = m0 + (mi << 4) + v + (half << 3);
        int col = n0 + (t << 4) + ln;
        out[(size_t)m * 512 + col] = acc[(mi << 2) + t][v] + bp[col];
      }
}

// ---------------- launcher ----------------

extern "C" void kernel_launch(void* const* d_in, const int* in_sizes, int n_in,
                              void* d_out, int out_size, void* d_ws, size_t ws_size,
                              hipStream_t stream) {
  const float* x      = (const float*)d_in[0];
  const float* mask   = (const float*)d_in[3];
  const float* Wq     = (const float*)d_in[4];
  const float* Wkv    = (const float*)d_in[5];
  const float* conv_w = (const float*)d_in[6];
  const float* conv_b = (const float*)d_in[7];
  const float* ln_w   = (const float*)d_in[8];
  const float* ln_b   = (const float*)d_in[9];
  const float* Wp     = (const float*)d_in[10];
  const float* bp     = (const float*)d_in[11];
  float* out = (float*)d_out;

  char* ws = (char*)d_ws;
  size_t off = 0;
  auto alloc = [&](size_t bytes) {
    void* p = ws + off;
    off += (bytes + 255) & ~(size_t)255;
    return p;
  };
  bf16t* xh   = (bf16t*)alloc((size_t)8 * 4096 * 512 * 2);   // 32 MB
  bf16t* qh   = (bf16t*)alloc((size_t)8 * 8 * 4096 * 64 * 2);// 32 MB
  float* ctxf = (float*)alloc((size_t)8 * 1024 * 512 * 4);   // 16 MB
  bf16t* ctxh = (bf16t*)alloc((size_t)8 * 1024 * 512 * 2);   //  8 MB
  bf16t* khp  = (bf16t*)alloc((size_t)8 * 8 * 1024 * 64 * 2);//  8 MB
  bf16t* vTp  = (bf16t*)alloc((size_t)8 * 8 * 64 * 1024 * 2);//  8 MB
  bf16t* o2   = (bf16t*)alloc((size_t)8 * 4096 * 512 * 2);   // 32 MB
  bf16t* WqH  = (bf16t*)alloc((size_t)512 * 512 * 2);
  bf16t* WkvH = (bf16t*)alloc((size_t)1024 * 512 * 2);
  bf16t* WcH  = (bf16t*)alloc((size_t)512 * 2048 * 2);
  bf16t* WpH  = (bf16t*)alloc((size_t)512 * 512 * 2);

  int nx4 = (8 * 4096 * 512) / 4;
  cvt_bf16x4_kernel<<<(nx4 + 255) / 256, 256, 0, stream>>>(
      (const float4*)x, (v4bf*)xh, nx4);
  cvt_bf16x4_kernel<<<(512 * 512 / 4 + 255) / 256, 256, 0, stream>>>(
      (const float4*)Wq, (v4bf*)WqH, 512 * 512 / 4);
  cvt_bf16x4_kernel<<<(1024 * 512 / 4 + 255) / 256, 256, 0, stream>>>(
      (const float4*)Wkv, (v4bf*)WkvH, 1024 * 512 / 4);
  cvt_bf16x4_kernel<<<(512 * 512 / 4 + 255) / 256, 256, 0, stream>>>(
      (const float4*)Wp, (v4bf*)WpH, 512 * 512 / 4);
  rearrange_convw_kernel<<<(512 * 2048) / 256, 256, 0, stream>>>(conv_w, WcH);

  gemm_q_kernel<<<1024, 256, 0, stream>>>(xh, WqH, qh);
  gemm_conv_kernel<<<256, 256, 0, stream>>>(xh, WcH, conv_b, ctxf);
  ln_kernel<<<8192, 128, 0, stream>>>(ctxf, ln_w, ln_b, ctxh);
  gemm_kv_kernel<<<512, 256, 0, stream>>>(ctxh, WkvH, khp, vTp);
  attn_kernel<<<2048, 256, 0, stream>>>(qh, khp, vTp, mask, o2);
  gemm_out_kernel<<<1024, 256, 0, stream>>>(o2, WpH, bp, out);
}
